// DeepGraphGOModel_29377576305015
// MI455X (gfx1250) — matrix operand order repaired and checked by reading, not compiled, not run on hardware
//
#include <hip/hip_runtime.h>
#include <hip/hip_bf16.h>
#include <math.h>

typedef __attribute__((ext_vector_type(16))) _Float16 v16h;
typedef __attribute__((ext_vector_type(8)))  _Float16 v8h;
typedef __attribute__((ext_vector_type(8)))  float    v8f;
typedef __attribute__((ext_vector_type(4)))  int      i32x4;

// gfx1250 async global->LDS path (ASYNCcnt), if the toolchain exposes it.
#if defined(__has_builtin)
#  if __has_builtin(__builtin_amdgcn_global_load_async_to_lds_b128) && \
      __has_builtin(__builtin_amdgcn_s_wait_asynccnt)
#    define USE_ASYNC_LDS 1
#  endif
#endif
#ifndef USE_ASYNC_LDS
#  define USE_ASYNC_LDS 0
#endif

#if USE_ASYNC_LDS
typedef __attribute__((address_space(1))) i32x4* gbl_i32x4_ptr;
typedef __attribute__((address_space(3))) i32x4* lds_i32x4_ptr;
#endif

// ---------------------------------------------------------------------------
// fp32 [K][N] -> f16 transposed [N][K], 32x32 LDS tile transpose
// ---------------------------------------------------------------------------
__global__ __launch_bounds__(256) void convert_transpose_w(
    const float* __restrict__ W, _Float16* __restrict__ WT, int K, int N) {
  __shared__ float tile[32][33];
  const int tx = threadIdx.x & 31;
  const int ty = threadIdx.x >> 5;  // 0..7
  const int k0 = blockIdx.y * 32;
  const int n0 = blockIdx.x * 32;
#pragma unroll
  for (int r = ty; r < 32; r += 8) {
    int k = k0 + r, n = n0 + tx;
    tile[r][tx] = (k < K && n < N) ? W[(size_t)k * N + n] : 0.0f;
  }
  __syncthreads();
#pragma unroll
  for (int r = ty; r < 32; r += 8) {
    int n = n0 + r, k = k0 + tx;
    if (n < N && k < K) WT[(size_t)n * K + k] = (_Float16)tile[tx][r];
  }
}

// ---------------------------------------------------------------------------
// Tiled WMMA GEMM:
//   C[M,N] = act( (rowScale? diag(rowScale):I) * A[M,K] @ BT[N,K]^T + bias )
// A fp32 (converted to f16 + optional row scale on the LDS staging path),
// BT is pre-converted f16, transposed [N][K] -> b128 copies into LDS
// (async global->LDS via GLOBAL_LOAD_ASYNC_TO_LDS_B128 when available).
// Block tile 128x128, K-step 64, 256 threads = 8 waves (wave32),
// each wave computes 32x64 = 2x(2x4) v_wmma_f32_16x16x32_f16 per K-tile.
// ACT: 0 = identity, 1 = relu, 2 = sigmoid
// ---------------------------------------------------------------------------
template <int ACT>
__global__ __launch_bounds__(256) void gemm_wmma_f16(
    const float* __restrict__ A, const _Float16* __restrict__ BT,
    const float* __restrict__ bias, const float* __restrict__ rowScale,
    float* __restrict__ C, int M, int N, int K) {
  __shared__ _Float16 lsA[128][72];  // [m][k0..63], 144B row pitch (bank-clean)
  __shared__ _Float16 lsB[128][72];  // [n][k0..63]

  const int tid    = threadIdx.x;
  const int lane   = tid & 31;
  const int wid    = tid >> 5;   // 0..7
  const int waveM  = wid & 3;    // 4 row-groups of 32
  const int waveN  = wid >> 2;   // 2 col-groups of 64
  const int half   = lane >> 4;  // which K-half of the fragment this lane holds
  const int l16    = lane & 15;

  const int blockN = blockIdx.x * 128;
  const int blockM = blockIdx.y * 128;

  v8f acc[2][4];
#pragma unroll
  for (int mi = 0; mi < 2; ++mi)
#pragma unroll
    for (int ni = 0; ni < 4; ++ni) acc[mi][ni] = (v8f)0.0f;

  const int numK = K >> 6;  // K is a multiple of 64 for every call here
  for (int kt = 0; kt < numK; ++kt) {
    const int k0 = kt << 6;

    // ---- stage B tile (128n x 64k f16): b128 copies, no math ----
#pragma unroll
    for (int i = 0; i < 4; ++i) {
      int idx = tid + i * 256;     // 0..1023 v8h slots
      int n   = idx >> 3;          // 8 v8h per 64-wide row
      int k8  = (idx & 7) << 3;
      int gn  = blockN + n;
      if (gn < N) {
        const _Float16* bp = BT + (size_t)gn * K + (k0 + k8);
#if USE_ASYNC_LDS
        __builtin_amdgcn_global_load_async_to_lds_b128(
            (gbl_i32x4_ptr)(const void*)bp,
            (lds_i32x4_ptr)(void*)&lsB[n][k8], 0, 0);
#else
        *(v8h*)&lsB[n][k8] = *(const v8h*)bp;
#endif
        if (k0 + 64 < K) __builtin_prefetch(bp + 64, 0, 3);  // global_prefetch_b8
      } else {
        *(v8h*)&lsB[n][k8] = (v8h)(_Float16)0.0f;
      }
    }

    // ---- stage A tile (128x64 fp32 -> f16), optional per-row scale ----
#pragma unroll
    for (int i = 0; i < 8; ++i) {
      int idx = tid + i * 256;     // 0..2047 float4 slots
      int row = idx >> 4;          // 16 float4 per 64-wide row
      int c4  = (idx & 15) << 2;
      int gr  = blockM + row;
      float4 v = make_float4(0.f, 0.f, 0.f, 0.f);
      float  sc = 1.0f;
      if (gr < M) {
        const float* ap = A + (size_t)gr * K + (k0 + c4);
        v = *(const float4*)ap;
        if (rowScale) sc = rowScale[gr];
        if (k0 + 64 < K) __builtin_prefetch(ap + 64, 0, 3);
      }
      lsA[row][c4 + 0] = (_Float16)(v.x * sc);
      lsA[row][c4 + 1] = (_Float16)(v.y * sc);
      lsA[row][c4 + 2] = (_Float16)(v.z * sc);
      lsA[row][c4 + 3] = (_Float16)(v.w * sc);
    }

#if USE_ASYNC_LDS
    __builtin_amdgcn_s_wait_asynccnt(0);
#endif
    __syncthreads();

    // ---- two K=32 sub-steps: 16 WMMAs per wave per staged tile ----
#pragma unroll
    for (int ks = 0; ks < 2; ++ks) {
      const int kb = ks * 32;
      // A 16x32 frag: lane<16 holds row l16, K {kb..kb+7, kb+16..kb+23}
      v16h afrag[2];
#pragma unroll
      for (int mi = 0; mi < 2; ++mi) {
        int row = waveM * 32 + mi * 16 + l16;
        int ka  = kb + half * 8;
        v8h lo = *(const v8h*)&lsA[row][ka];
        v8h hi = *(const v8h*)&lsA[row][ka + 16];
        v16h a;
#pragma unroll
        for (int j = 0; j < 8; ++j) { a[j] = lo[j]; a[8 + j] = hi[j]; }
        afrag[mi] = a;
      }
      // B 32x16 frag: lane<16 holds col l16, K {kb..kb+7, kb+16..kb+23}
      v16h bfrag[4];
#pragma unroll
      for (int ni = 0; ni < 4; ++ni) {
        int n   = waveN * 64 + ni * 16 + l16;
        int kc  = kb + half * 8;
        v8h lo = *(const v8h*)&lsB[n][kc];
        v8h hi = *(const v8h*)&lsB[n][kc + 16];
        v16h b;
#pragma unroll
        for (int j = 0; j < 8; ++j) { b[j] = lo[j]; b[8 + j] = hi[j]; }
        bfrag[ni] = b;
      }
#pragma unroll
      for (int mi = 0; mi < 2; ++mi)
#pragma unroll
        for (int ni = 0; ni < 4; ++ni)
          acc[mi][ni] = __builtin_amdgcn_wmma_f32_16x16x32_f16(
              false, afrag[mi], false, bfrag[ni], (short)0, acc[mi][ni], false, false);
    }

    __syncthreads();
  }

  // ---- epilogue: bias + activation + guarded store (coalesced per half-wave) ----
#pragma unroll
  for (int mi = 0; mi < 2; ++mi) {
#pragma unroll
    for (int ni = 0; ni < 4; ++ni) {
      int col = blockN + waveN * 64 + ni * 16 + l16;
      if (col >= N) continue;
      float bv = bias[col];
      int rowBase = blockM + waveM * 32 + mi * 16 + half * 8;
#pragma unroll
      for (int r = 0; r < 8; ++r) {
        int row = rowBase + r;
        if (row < M) {
          float v = acc[mi][ni][r] + bv;
          if (ACT == 1) v = fmaxf(v, 0.0f);
          if (ACT == 2) v = 1.0f / (1.0f + expf(-v));
          C[(size_t)row * N + col] = v;
        }
      }
    }
  }
}

// ---------------------------------------------------------------------------
// Degree accumulation: deg_out[src[e]] += 1, deg_in[dst[e]] += 1
// ---------------------------------------------------------------------------
__global__ __launch_bounds__(256) void degree_kernel(
    const int* __restrict__ src, const int* __restrict__ dst,
    float* __restrict__ degOut, float* __restrict__ degIn, int E) {
  int e = blockIdx.x * blockDim.x + threadIdx.x;
  if (e < E) {
    atomicAdd(&degOut[src[e]], 1.0f);
    atomicAdd(&degIn[dst[e]], 1.0f);
  }
}

// v = rsqrt(max(v, 1)) in place over the 4 degree arrays at once
__global__ __launch_bounds__(256) void finish_scales_kernel(float* __restrict__ v, int n) {
  int i = blockIdx.x * blockDim.x + threadIdx.x;
  if (i < n) v[i] = rsqrtf(fmaxf(v[i], 1.0f));
}

// ---------------------------------------------------------------------------
// Edge scatter: agg[dst[e], :] += x[src[e], :] * sOut[src[e]]
// One wave32 per edge, float4 gathers + per-element f32 atomics (L2-resident).
// ---------------------------------------------------------------------------
__global__ __launch_bounds__(256) void spmm_scatter_kernel(
    const float* __restrict__ x, const int* __restrict__ src,
    const int* __restrict__ dst, const float* __restrict__ sOut,
    float* __restrict__ agg, int E, int H) {
  int wave = blockIdx.x * (blockDim.x >> 5) + (threadIdx.x >> 5);
  if (wave >= E) return;
  int lane = threadIdx.x & 31;
  int s = src[wave];
  int d = dst[wave];
  float sc = sOut[s];
  const float4* xr = (const float4*)(x + (size_t)s * H);
  float* ar = agg + (size_t)d * H;
  int nvec = H >> 2;
  for (int c = lane; c < nvec; c += 32) {
    float4 v = xr[c];
    int b = c << 2;
    atomicAdd(ar + b + 0, v.x * sc);
    atomicAdd(ar + b + 1, v.y * sc);
    atomicAdd(ar + b + 2, v.z * sc);
    atomicAdd(ar + b + 3, v.w * sc);
  }
}

// ---------------------------------------------------------------------------
extern "C" void kernel_launch(void* const* d_in, const int* in_sizes, int n_in,
                              void* d_out, int out_size, void* d_ws, size_t ws_size,
                              hipStream_t stream) {
  const float* feat = (const float*)d_in[0];
  const float* W1   = (const float*)d_in[1];
  const float* b1   = (const float*)d_in[2];
  const float* Wc1  = (const float*)d_in[3];
  const float* bc1  = (const float*)d_in[4];
  const float* Wc2  = (const float*)d_in[5];
  const float* bc2  = (const float*)d_in[6];
  const float* W2   = (const float*)d_in[7];
  const float* b2   = (const float*)d_in[8];
  const int* src1   = (const int*)d_in[9];
  const int* dst1   = (const int*)d_in[10];
  const int* src2   = (const int*)d_in[11];
  const int* dst2   = (const int*)d_in[12];

  const int H   = in_sizes[2];        // 1024
  const int Kin = in_sizes[1] / H;    // 2048
  const int Nn  = in_sizes[0] / Kin;  // 20000 nodes
  const int NG  = in_sizes[8];        // 5000
  const int E1  = in_sizes[9];
  const int E2  = in_sizes[11];

  // ---- workspace layout ----
  float* ws    = (float*)d_ws;
  float* sOut1 = ws;
  float* sIn1  = ws + (size_t)Nn;
  float* sOut2 = ws + (size_t)2 * Nn;
  float* sIn2  = ws + (size_t)3 * Nn;
  size_t off   = (((size_t)4 * Nn) + 63) & ~(size_t)63;   // in floats
  float* buf0  = ws + off;  off += (size_t)Nn * H;        // [Nn, H] fp32
  float* buf1  = ws + off;  off += (size_t)Nn * H;        // [Nn, H] fp32
  _Float16* hbase = (_Float16*)(ws + off);
  size_t hoff = 0;
  _Float16* wt1  = hbase + hoff; hoff += (size_t)Kin * H;  // [H][Kin]  f16
  _Float16* wtc1 = hbase + hoff; hoff += (size_t)H * H;    // [H][H]
  _Float16* wtc2 = hbase + hoff; hoff += (size_t)H * H;    // [H][H]
  _Float16* wt2  = hbase + hoff; hoff += (size_t)H * NG;   // [NG][H]

  dim3 blk(256);
  auto grid2d = [](int M, int N) {
    return dim3((unsigned)((N + 127) / 128), (unsigned)((M + 127) / 128));
  };
  auto gridT = [](int K, int N) {
    return dim3((unsigned)((N + 31) / 32), (unsigned)((K + 31) / 32));
  };

  // ---- 0) pre-convert + transpose all weights to f16 [N][K] ----
  convert_transpose_w<<<gridT(Kin, H), blk, 0, stream>>>(W1, wt1, Kin, H);
  convert_transpose_w<<<gridT(H, H),   blk, 0, stream>>>(Wc1, wtc1, H, H);
  convert_transpose_w<<<gridT(H, H),   blk, 0, stream>>>(Wc2, wtc2, H, H);
  convert_transpose_w<<<gridT(H, NG),  blk, 0, stream>>>(W2, wt2, H, NG);

  // ---- 1) x = relu(features @ W1 + b1) ----
  gemm_wmma_f16<1><<<grid2d(Nn, H), blk, 0, stream>>>(feat, wt1, b1, nullptr, buf0, Nn, H, Kin);

  // ---- 2) degrees and normalization scales for both graphs ----
  (void)hipMemsetAsync(ws, 0, (size_t)4 * Nn * sizeof(float), stream);
  degree_kernel<<<(E1 + 255) / 256, blk, 0, stream>>>(src1, dst1, sOut1, sIn1, E1);
  degree_kernel<<<(E2 + 255) / 256, blk, 0, stream>>>(src2, dst2, sOut2, sIn2, E2);
  finish_scales_kernel<<<(4 * Nn + 255) / 256, blk, 0, stream>>>(ws, 4 * Nn);

  // ---- 3) GraphConv 1: agg = scatter(x * dOut^-1/2); x = (agg * dIn^-1/2) @ Wc1 + bc1 ----
  (void)hipMemsetAsync(buf1, 0, (size_t)Nn * H * sizeof(float), stream);
  spmm_scatter_kernel<<<(E1 + 7) / 8, blk, 0, stream>>>(buf0, src1, dst1, sOut1, buf1, E1, H);
  gemm_wmma_f16<0><<<grid2d(Nn, H), blk, 0, stream>>>(buf1, wtc1, bc1, sIn1, buf0, Nn, H, H);

  // ---- 4) GraphConv 2 ----
  (void)hipMemsetAsync(buf1, 0, (size_t)Nn * H * sizeof(float), stream);
  spmm_scatter_kernel<<<(E2 + 7) / 8, blk, 0, stream>>>(buf0, src2, dst2, sOut2, buf1, E2, H);
  gemm_wmma_f16<0><<<grid2d(Nn, H), blk, 0, stream>>>(buf1, wtc2, bc2, sIn2, buf0, Nn, H, H);

  // ---- 5) logits = sigmoid(x @ W2 + b2) ----
  gemm_wmma_f16<2><<<grid2d(Nn, NG), blk, 0, stream>>>(buf0, wt2, b2, nullptr, (float*)d_out, Nn, NG, H);
}
